// GPT2Small_47785806135457
// MI455X (gfx1250) — matrix operand order repaired and checked by reading, not compiled, not run on hardware
//
#include <hip/hip_runtime.h>
#include <hip/hip_bf16.h>
#include <stdint.h>

// ---------------------------------------------------------------------------
// GPT-2 small forward on gfx1250 (CDNA5). All GEMMs use v_wmma_f32_16x16x32_bf16.
// Weights are streamed fp32 (memory-bound: ~0.5GB @ 23.3TB/s), converted to bf16
// once per block via cooperative LDS staging (8 waves share one W tile), with
// global_prefetch of the tile two K-steps ahead.
// ---------------------------------------------------------------------------

typedef __attribute__((ext_vector_type(16))) __bf16   v16bf;
typedef __attribute__((ext_vector_type(8)))  float    v8f;
typedef __attribute__((ext_vector_type(8)))  uint32_t v8u;

__device__ __forceinline__ float wave_sum_f(float v) {
#pragma unroll
  for (int off = 16; off > 0; off >>= 1) v += __shfl_xor(v, off, 32);
  return v;
}
__device__ __forceinline__ float wave_max_f(float v) {
#pragma unroll
  for (int off = 16; off > 0; off >>= 1) v = fmaxf(v, __shfl_xor(v, off, 32));
  return v;
}

// Exact v_cvt_pk_bf16_f32 (RNE): dst = {bf16(a) in [15:0], bf16(b) in [31:16]}.
__device__ __forceinline__ uint32_t pack2bf(float a, float b) {
  uint32_t r;
  asm("v_cvt_pk_bf16_f32 %0, %1, %2" : "=v"(r) : "v"(a), "v"(b));
  return r;
}

__device__ __forceinline__ void pack8(v16bf& f, int base, float4 a, float4 b) {
  f[base + 0] = (__bf16)a.x; f[base + 1] = (__bf16)a.y;
  f[base + 2] = (__bf16)a.z; f[base + 3] = (__bf16)a.w;
  f[base + 4] = (__bf16)b.x; f[base + 5] = (__bf16)b.y;
  f[base + 6] = (__bf16)b.z; f[base + 7] = (__bf16)b.w;
}

// A fragment 16x32 bf16 (ISA 16-bit A layout):
// lane(ll,hi): elems 0..7  = A[m0+ll][k0 + hi*8 + 0..7]
//              elems 8..15 = A[m0+ll][k0 + 16 + hi*8 + 0..7]
__device__ __forceinline__ v16bf load_a_frag(const float* __restrict__ rowp,
                                             int k0, int hi) {
  const float4* p = reinterpret_cast<const float4*>(rowp + k0 + hi * 8);
  v16bf f;
  pack8(f, 0, p[0], p[1]);   // k0+hi*8 .. +7
  pack8(f, 8, p[4], p[5]);   // k0+16+hi*8 .. +7
  return f;
}

// Stage one 64(cols) x 32(K) W tile as pre-packed bf16 B-fragments into LDS.
// B = W^T (W row-major [N,K]); B-frag lane(ll,hi) elem e = W[n0+nt*16+ll][k0+hi*8+e] (e<8), [k0+16+hi*8+e-8] (e>=8).
// Packed-uint u (0..7) of that lane = W[n][k0+(u>>2)*16+hi*8+2(u&3)], [+1].
// LDS layout: lds[(nt*32 + hi*16 + ll)*8 + u].
// Thread t (0..255): n = n0 + (t&63); p = t>>6 -> hi = p>>1, half = p&1 (u = half*4..+3).
__device__ __forceinline__ void stage_w(const float* __restrict__ W, int N, int K,
                                        int n0, int k0, uint32_t* __restrict__ lds,
                                        int t) {
  const int nl   = t & 63;
  const int p    = t >> 6;
  const int hi   = p >> 1;
  const int half = p & 1;
  const int n    = n0 + nl;
  uint32_t u0, u1, u2, u3;
  if (n < N) {
    const float* src = W + (size_t)n * K + k0 + half * 16 + hi * 8;
    const float4* s4 = reinterpret_cast<const float4*>(src);
    float4 x0 = s4[0], x1 = s4[1];
    u0 = pack2bf(x0.x, x0.y); u1 = pack2bf(x0.z, x0.w);
    u2 = pack2bf(x1.x, x1.y); u3 = pack2bf(x1.z, x1.w);
    if (k0 + 64 < K) __builtin_prefetch(src + 64, 0, 3);  // global_prefetch, 2 steps ahead
  } else {
    u0 = u1 = u2 = u3 = 0u;
  }
  const int nt   = nl >> 4;
  const int lane = hi * 16 + (nl & 15);
  uint4* dst = reinterpret_cast<uint4*>(lds + ((nt * 32 + lane) * 8) + half * 4);
  *dst = uint4{u0, u1, u2, u3};
}

__device__ __forceinline__ float gelu_exact(float x) {
  return 0.5f * x * (1.0f + erff(x * 0.70710678118654752f));
}

// C = act(A[M,K] @ W[N,K]^T + bias) + resid   (bias/resid nullable, act: 0=none 1=gelu)
// Block = 8 waves: 128 rows x 64 cols. W tile staged in LDS (double-buffered) by all
// 256 threads; each wave reads its 4 B-frags via ds_load_b128.
__global__ void __launch_bounds__(256)
gemm_bf16_wmma(const float* __restrict__ A, const float* __restrict__ W,
               const float* __restrict__ bias, const float* __restrict__ resid,
               float* __restrict__ C, int M, int N, int K, int act) {
  __shared__ uint32_t wtile[2][4 * 32 * 8];   // 2 x 4KB
  const int tid  = threadIdx.x;
  const int wid  = tid >> 5;
  const int lane = tid & 31;
  const int ll   = lane & 15;
  const int hi   = lane >> 4;
  int m0 = (blockIdx.y * 8 + wid) * 16;
  if (m0 + 16 > M) m0 = (M > 16) ? (M - 16) : 0;   // clamp (no divergent barrier)
  const int n0 = blockIdx.x * 64;

  const float* abase = A + (size_t)(m0 + ll) * K;

  stage_w(W, N, K, n0, 0, wtile[0], tid);
  __syncthreads();

  v8f c0 = {}, c1 = {}, c2 = {}, c3 = {};
  for (int k0 = 0; k0 < K; k0 += 32) {
    const int cur = (k0 >> 5) & 1;
    if (k0 + 32 < K) stage_w(W, N, K, n0, k0 + 32, wtile[cur ^ 1], tid);

    const v16bf a = load_a_frag(abase, k0, hi);
    const uint32_t* bb = &wtile[cur][0];
    const v16bf b0 = __builtin_bit_cast(v16bf, *reinterpret_cast<const v8u*>(bb + (0 * 32 + lane) * 8));
    const v16bf b1 = __builtin_bit_cast(v16bf, *reinterpret_cast<const v8u*>(bb + (1 * 32 + lane) * 8));
    const v16bf b2 = __builtin_bit_cast(v16bf, *reinterpret_cast<const v8u*>(bb + (2 * 32 + lane) * 8));
    const v16bf b3 = __builtin_bit_cast(v16bf, *reinterpret_cast<const v8u*>(bb + (3 * 32 + lane) * 8));
    c0 = __builtin_amdgcn_wmma_f32_16x16x32_bf16(false, a, false, b0, (short)0, c0, false, false);
    c1 = __builtin_amdgcn_wmma_f32_16x16x32_bf16(false, a, false, b1, (short)0, c1, false, false);
    c2 = __builtin_amdgcn_wmma_f32_16x16x32_bf16(false, a, false, b2, (short)0, c2, false, false);
    c3 = __builtin_amdgcn_wmma_f32_16x16x32_bf16(false, a, false, b3, (short)0, c3, false, false);
    __syncthreads();
  }

  // C/D layout: VGPR r -> row m0+r (lanes 0-15) / m0+r+8 (lanes 16-31).
#pragma unroll
  for (int r = 0; r < 8; ++r) {
    const int row = m0 + r + hi * 8;
    if (row >= M) continue;
    float vals[4] = { c0[r], c1[r], c2[r], c3[r] };
#pragma unroll
    for (int t = 0; t < 4; ++t) {
      const int cc = n0 + t * 16 + ll;
      if (cc < N) {
        float v = vals[t];
        if (bias)   v += bias[cc];
        if (act == 1) v = gelu_exact(v);
        const size_t idx = (size_t)row * N + cc;
        if (resid)  v += resid[idx];
        C[idx] = v;
      }
    }
  }
}

// Token + positional embedding gather.
__global__ void embed_kernel(const int* __restrict__ ids, const float* __restrict__ tok,
                             const float* __restrict__ pos, float* __restrict__ x,
                             int BT, int T, int D) {
  int idx = blockIdx.x * blockDim.x + threadIdx.x;
  if (idx >= BT * D) return;
  int bt = idx / D, d = idx - bt * D;
  int t = bt % T;
  x[idx] = tok[(size_t)ids[bt] * D + d] + pos[(size_t)t * D + d];
}

// LayerNorm: one wave per row of length D.
__global__ void __launch_bounds__(256)
layernorm_kernel(const float* __restrict__ x, const float* __restrict__ w,
                 const float* __restrict__ b, float* __restrict__ y, int M, int D) {
  const int wid = threadIdx.x >> 5, lane = threadIdx.x & 31;
  const int row = blockIdx.x * 8 + wid;
  if (row >= M) return;
  const float* xr = x + (size_t)row * D;
  float s = 0.f, sq = 0.f;
  for (int i = lane; i < D; i += 32) { float v = xr[i]; s += v; sq += v * v; }
  s  = wave_sum_f(s);
  sq = wave_sum_f(sq);
  const float mean = s / (float)D;
  const float var  = sq / (float)D - mean * mean;
  const float rstd = rsqrtf(var + 1e-5f);
  float* yr = y + (size_t)row * D;
  for (int i = lane; i < D; i += 32) yr[i] = (xr[i] - mean) * rstd * w[i] + b[i];
}

// Causal attention over packed qkv[B*T, 3*C]; one wave per (b,h,qt) query row.
#define T_SEQ 512
#define HD    64
#define NH    12
#define C3    2304
__global__ void __launch_bounds__(256)
attn_kernel(const float* __restrict__ qkv, float* __restrict__ o, int Bn) {
  __shared__ float sc[8][T_SEQ];
  __shared__ float qs[8][HD];
  const int wid  = threadIdx.x >> 5;
  const int lane = threadIdx.x & 31;
  const int gw   = blockIdx.x * 8 + wid;
  if (gw >= Bn * NH * T_SEQ) return;   // wave-uniform exit; no cross-wave barriers used
  const int qt = gw % T_SEQ;
  const int bh = gw / T_SEQ;
  const int h  = bh % NH;
  const int b  = bh / NH;

  const float* qp    = qkv + (size_t)(b * T_SEQ + qt) * C3 + h * HD;   // q section
  const float* kbase = qkv + (size_t)b * T_SEQ * C3 + 768 + h * HD;    // k section
  const float* vbase = kbase + 768;                                     // v section

  qs[wid][lane]      = qp[lane];
  qs[wid][lane + 32] = qp[lane + 32];

  // scores + running max (causal: kt <= qt only)
  float lmax = -3.0e38f;
  for (int kt = lane; kt <= qt; kt += 32) {
    const float4* kp = reinterpret_cast<const float4*>(kbase + (size_t)kt * C3);
    float dot = 0.f;
#pragma unroll
    for (int j = 0; j < 16; ++j) {
      float4 kv = kp[j];
      dot += kv.x * qs[wid][4 * j + 0] + kv.y * qs[wid][4 * j + 1] +
             kv.z * qs[wid][4 * j + 2] + kv.w * qs[wid][4 * j + 3];
    }
    dot *= 0.125f;                    // hd^-0.5, hd = 64
    sc[wid][kt] = dot;
    lmax = fmaxf(lmax, dot);
  }
  const float m = wave_max_f(lmax);
  float lsum = 0.f;
  for (int kt = lane; kt <= qt; kt += 32) {
    float e = __expf(sc[wid][kt] - m);
    sc[wid][kt] = e;
    lsum += e;
  }
  const float inv = 1.0f / wave_sum_f(lsum);

  // o[qt,:] = sum_kt p[kt] * v[kt,:]; each lane owns 2 of the 64 dims
  const int d0 = lane * 2;
  float a0 = 0.f, a1 = 0.f;
  for (int kt = 0; kt <= qt; ++kt) {
    const float  p  = sc[wid][kt];
    const float* vp = vbase + (size_t)kt * C3;
    a0 += p * vp[d0];
    a1 += p * vp[d0 + 1];
  }
  float* op = o + (size_t)(b * T_SEQ + qt) * 768 + h * HD;
  op[d0]     = a0 * inv;
  op[d0 + 1] = a1 * inv;
}

extern "C" void kernel_launch(void* const* d_in, const int* in_sizes, int n_in,
                              void* d_out, int out_size, void* d_ws, size_t ws_size,
                              hipStream_t stream) {
  (void)in_sizes; (void)n_in; (void)out_size; (void)ws_size;
  const int*   ids   = (const int*)  d_in[0];
  const float* tok   = (const float*)d_in[1];
  const float* pos   = (const float*)d_in[2];
  const float* ln1w  = (const float*)d_in[3];
  const float* ln1b  = (const float*)d_in[4];
  const float* qkvw  = (const float*)d_in[5];
  const float* qkvb  = (const float*)d_in[6];
  const float* projw = (const float*)d_in[7];
  const float* projb = (const float*)d_in[8];
  const float* ln2w  = (const float*)d_in[9];
  const float* ln2b  = (const float*)d_in[10];
  const float* fc1w  = (const float*)d_in[11];
  const float* fc1b  = (const float*)d_in[12];
  const float* fc2w  = (const float*)d_in[13];
  const float* fc2b  = (const float*)d_in[14];
  const float* lnfw  = (const float*)d_in[15];
  const float* lnfb  = (const float*)d_in[16];

  const int Bn = 2, T = 512, D = 768, Dm = 3072, V = 50257, L = 12;
  const int BT = Bn * T;                 // 1024

  float* ws = (float*)d_ws;
  const size_t X = (size_t)BT * D;       // 786432 floats
  float* x    = ws;                      // [BT, D]
  float* h    = ws + X;                  // [BT, D]
  float* qkv  = ws + 2 * X;              // [BT, 3D]  (3X)
  float* oatt = ws + 5 * X;              // [BT, D]
  float* mlp  = ws + 6 * X;              // [BT, Dm]  (4X)  -> total 10X ~ 31.5 MB

  dim3 blk(256);
  embed_kernel<<<(BT * D + 255) / 256, blk, 0, stream>>>(ids, tok, pos, x, BT, T, D);

  const dim3 gQKV((3 * D + 63) / 64, (BT + 127) / 128);
  const dim3 gPRJ((D + 63) / 64,     (BT + 127) / 128);
  const dim3 gFC1((Dm + 63) / 64,    (BT + 127) / 128);
  const dim3 gLM((V + 63) / 64,      (BT + 127) / 128);

  for (int l = 0; l < L; ++l) {
    layernorm_kernel<<<BT / 8, blk, 0, stream>>>(x, ln1w + l * D, ln1b + l * D, h, BT, D);
    gemm_bf16_wmma<<<gQKV, blk, 0, stream>>>(h, qkvw + (size_t)l * 3 * D * D,
                                             qkvb + (size_t)l * 3 * D, nullptr,
                                             qkv, BT, 3 * D, D, 0);
    attn_kernel<<<(Bn * NH * T_SEQ) / 8, blk, 0, stream>>>(qkv, oatt, Bn);
    gemm_bf16_wmma<<<gPRJ, blk, 0, stream>>>(oatt, projw + (size_t)l * D * D,
                                             projb + (size_t)l * D, x,
                                             x, BT, D, D, 0);
    layernorm_kernel<<<BT / 8, blk, 0, stream>>>(x, ln2w + l * D, ln2b + l * D, h, BT, D);
    gemm_bf16_wmma<<<gFC1, blk, 0, stream>>>(h, fc1w + (size_t)l * Dm * D,
                                             fc1b + (size_t)l * Dm, nullptr,
                                             mlp, BT, Dm, D, 1 /*gelu*/);
    gemm_bf16_wmma<<<gPRJ, blk, 0, stream>>>(mlp, fc2w + (size_t)l * D * Dm,
                                             fc2b + (size_t)l * D, x,
                                             x, BT, D, Dm, 0);
  }

  layernorm_kernel<<<BT / 8, blk, 0, stream>>>(x, lnfw, lnfb, h, BT, D);
  gemm_bf16_wmma<<<gLM, blk, 0, stream>>>(h, tok, nullptr, nullptr,
                                          (float*)d_out, BT, V, D, 0);
}
